// EpipolarAttention_79645873537735
// MI455X (gfx1250) — matrix operand order repaired and checked by reading, not compiled, not run on hardware
//
#include <hip/hip_runtime.h>
#include <hip/hip_bf16.h>

typedef __attribute__((ext_vector_type(16))) _Float16 v16h;
typedef __attribute__((ext_vector_type(8)))  _Float16 v8h;
typedef __attribute__((ext_vector_type(4)))  _Float16 v4h;
typedef __attribute__((ext_vector_type(8)))  float    v8f;

#define NB   16      // batches
#define HW   1024    // 32*32 pixels
#define NC   512     // channels

// ---------------- 3x3 helpers ----------------
__device__ inline void mat3_mul(const float* A, const float* Bm, float* Cm) {
  for (int i = 0; i < 3; i++)
    for (int j = 0; j < 3; j++)
      Cm[i*3+j] = A[i*3+0]*Bm[0+j] + A[i*3+1]*Bm[3+j] + A[i*3+2]*Bm[6+j];
}

__device__ inline void mat3_inv(const float* M, float* Inv) {
  float a=M[0],b=M[1],c=M[2],d=M[3],e=M[4],f=M[5],g=M[6],h=M[7],i=M[8];
  float A_ =  e*i - f*h;
  float B_ = -(d*i - f*g);
  float C_ =  d*h - e*g;
  float det = a*A_ + b*B_ + c*C_;
  float r = 1.0f / det;
  Inv[0]=A_*r;        Inv[1]=(c*h-b*i)*r; Inv[2]=(b*f-c*e)*r;
  Inv[3]=B_*r;        Inv[4]=(a*i-c*g)*r; Inv[5]=(c*d-a*f)*r;
  Inv[6]=C_*r;        Inv[7]=(b*g-a*h)*r; Inv[8]=(a*e-b*d)*r;
}

// smallest-eigenvalue eigenvector of symmetric 3x3 via cyclic Jacobi
__device__ void sym3_min_eigvec(const float* A0, float* v3) {
  float A[9]; for (int i=0;i<9;i++) A[i]=A0[i];
  float V[9] = {1,0,0, 0,1,0, 0,0,1};
  for (int sweep = 0; sweep < 12; sweep++) {
    for (int pass = 0; pass < 3; pass++) {
      int p = (pass == 2) ? 1 : 0;
      int q = (pass == 0) ? 1 : 2;
      float apq = A[p*3+q];
      if (fabsf(apq) < 1e-30f) continue;
      float theta = (A[q*3+q] - A[p*3+p]) / (2.0f * apq);
      float tt = 1.0f / (fabsf(theta) + sqrtf(theta*theta + 1.0f));
      if (theta < 0.0f) tt = -tt;
      float cth = 1.0f / sqrtf(tt*tt + 1.0f);
      float sth = tt * cth;
      for (int k = 0; k < 3; k++) {              // A = A * J
        float akp = A[k*3+p], akq = A[k*3+q];
        A[k*3+p] = cth*akp - sth*akq;
        A[k*3+q] = sth*akp + cth*akq;
      }
      for (int k = 0; k < 3; k++) {              // A = J^T * A
        float apk = A[p*3+k], aqk = A[q*3+k];
        A[p*3+k] = cth*apk - sth*aqk;
        A[q*3+k] = sth*apk + cth*aqk;
      }
      for (int k = 0; k < 3; k++) {              // V = V * J
        float vkp = V[k*3+p], vkq = V[k*3+q];
        V[k*3+p] = cth*vkp - sth*vkq;
        V[k*3+q] = sth*vkp + cth*vkq;
      }
    }
  }
  int mi = 0; float mv = A[0];
  if (A[4] < mv) { mv = A[4]; mi = 1; }
  if (A[8] < mv) { mv = A[8]; mi = 2; }
  v3[0] = V[0*3+mi]; v3[1] = V[1*3+mi]; v3[2] = V[2*3+mi];
}

// ---------------- kernel 1: F matrix + per-line affine coefficients ----------------
// d[b,i,k] = | px_i*alpha[b,k] + py_i*beta[b,k] + gamma[b,k] |
__global__ void k_setup(const float* __restrict__ K1, const float* __restrict__ K2,
                        const float* __restrict__ R,  const float* __restrict__ t,
                        float4* __restrict__ coefs) {
  __shared__ float Fs[9];
  int b = blockIdx.x;
  if (threadIdx.x == 0) {
    float tx = t[b*3+0], ty = t[b*3+1], tz = t[b*3+2];
    float skew[9] = {0.f,-tz, ty,  tz, 0.f,-tx,  -ty, tx, 0.f};
    float Rm[9], K1m[9], K2m[9];
    for (int i=0;i<9;i++){ Rm[i]=R[b*9+i]; K1m[i]=K1[b*9+i]; K2m[i]=K2[b*9+i]; }
    float E[9]; mat3_mul(skew, Rm, E);
    float A[9];                                  // E^T E
    for (int i=0;i<3;i++) for (int j=0;j<3;j++)
      A[i*3+j] = E[0+i]*E[0+j] + E[3+i]*E[3+j] + E[6+i]*E[6+j];
    float v3[3]; sym3_min_eigvec(A, v3);
    float Ev[3];
    for (int i=0;i<3;i++) Ev[i] = E[i*3+0]*v3[0] + E[i*3+1]*v3[1] + E[i*3+2]*v3[2];
    float Ep[9];                                 // rank-2 projection: E - (Ev) v3^T
    for (int i=0;i<3;i++) for (int j=0;j<3;j++) Ep[i*3+j] = E[i*3+j] - Ev[i]*v3[j];
    float M1[9], M2[9]; mat3_inv(K1m, M1); mat3_inv(K2m, M2);
    float M2T[9];
    for (int i=0;i<3;i++) for (int j=0;j<3;j++) M2T[i*3+j] = M2[j*3+i];
    float T1[9]; mat3_mul(M2T, Ep, T1);
    float F[9];  mat3_mul(T1, M1, F);
    for (int i=0;i<9;i++) Fs[i] = F[i];
  }
  __syncthreads();
  for (int k = threadIdx.x; k < HW; k += blockDim.x) {
    float px = (float)(k >> 5), py = (float)(k & 31);
    float la = Fs[0]*px + Fs[1]*py + Fs[2];
    float lb = Fs[3]*px + Fs[4]*py + Fs[5];
    float lc = Fs[6]*px + Fs[7]*py + Fs[8];
    float y0 = -lc / lb;
    float dy = la * 32.0f / lb;                  // (x0,y0)=(0,-c/b), dx=-W, dy=aW/b
    float rinv = 1.0f / sqrtf(1024.0f + dy*dy);  // 1/sqrt(dx^2+dy^2)
    coefs[b*HW + k] = make_float4(dy*rinv, 32.0f*rinv, -y0*32.0f*rinv, 0.0f);
  }
}

// ---------------- kernel 2: per-row softmax over k, P = exp(-s) in f16 ----------------
__global__ void __launch_bounds__(256) k_rowsoftmax(const float4* __restrict__ coefs,
                                                    _Float16* __restrict__ P) {
  __shared__ float red[256];
  int b = blockIdx.x >> 10;
  int i = blockIdx.x & 1023;
  float px = (float)(i >> 5), py = (float)(i & 31);
  int tid = threadIdx.x;
  float dv[4];
  for (int j = 0; j < 4; j++) {
    float4 cf = coefs[b*HW + tid + j*256];
    dv[j] = 5.0f * fabsf(px*cf.x + py*cf.y + cf.z);   // 5*d (the -0.1 shift cancels)
  }
  float m = fmaxf(fmaxf(dv[0], dv[1]), fmaxf(dv[2], dv[3]));
  red[tid] = m; __syncthreads();
  for (int s = 128; s > 0; s >>= 1) { if (tid < s) red[tid] = fmaxf(red[tid], red[tid+s]); __syncthreads(); }
  float rowmax = red[0]; __syncthreads();
  float e[4]; float acc = 0.0f;
  for (int j = 0; j < 4; j++) { e[j] = expf(dv[j] - rowmax); acc += e[j]; }
  red[tid] = acc; __syncthreads();
  for (int s = 128; s > 0; s >>= 1) { if (tid < s) red[tid] += red[tid+s]; __syncthreads(); }
  float rs = 1.0f / red[0];
  _Float16* Pr = P + (size_t)(b*HW + i) * HW;
  for (int j = 0; j < 4; j++) {
    float sv = e[j] * rs;                        // softmax_k(5d)
    Pr[tid + j*256] = (_Float16)expf(-sv);       // numerator of softmax_i(1 - s)
  }
}

// ---------------- kernel 3: column sums of P (partial + atomic) ----------------
__global__ void __launch_bounds__(256) k_colsum(const _Float16* __restrict__ P,
                                                float* __restrict__ colsum) {
  int blk = blockIdx.x;
  int ic = blk & 7;            // i chunk (8 x 128 rows)
  int kt = (blk >> 3) & 3;     // k tile  (4 x 256 cols)
  int b  = blk >> 5;
  int k = kt*256 + threadIdx.x;
  const _Float16* Pb = P + (size_t)b * HW * HW;
  float s = 0.0f;
  int i0 = ic * 128;
  for (int i = i0; i < i0 + 128; i++) s += (float)Pb[(size_t)i*HW + k];
  atomicAdd(&colsum[b*HW + k], s);
}

// ---------------- kernel 4: Vh[b][c][k] = f16( f_src / colsum ) ----------------
__global__ void __launch_bounds__(256) k_buildV(const float* __restrict__ f_src,
                                                const float* __restrict__ colsum,
                                                _Float16* __restrict__ Vh) {
  int idx4 = blockIdx.x * blockDim.x + threadIdx.x;  // float4 index
  int e = idx4 * 4;
  int k = e & 1023;
  int b = e >> 19;                                   // 512*1024 = 2^19 per batch
  float4 v  = ((const float4*)f_src)[idx4];
  float4 cs = *(const float4*)(colsum + (b << 10) + k);
  v4h o;
  o[0] = (_Float16)(v.x / cs.x); o[1] = (_Float16)(v.y / cs.y);
  o[2] = (_Float16)(v.z / cs.z); o[3] = (_Float16)(v.w / cs.w);
  *(v4h*)(Vh + e) = o;
}

// ---------------- kernel 5: O = P @ Vh^T  via v_wmma_f32_16x16x32_f16 ----------------
// Block: 256 threads = 8 waves (4 M-rows x 2 N-cols). Block tile 128x128.
// Wave tile 32(M) x 64(N) = 2x4 WMMA accumulators.
// k-loop unrolled x2 so each iteration consumes a full 128B cacheline per row.
__global__ void __launch_bounds__(256) k_gemm(const _Float16* __restrict__ P,
                                              const _Float16* __restrict__ Vh,
                                              float* __restrict__ out) {
  int b    = blockIdx.z;
  int lane = threadIdx.x & 31;
  int wave = threadIdx.x >> 5;
  int wm = wave & 3, wn = wave >> 2;
  int i0 = blockIdx.y * 128 + wm * 32;
  int c0 = blockIdx.x * 128 + wn * 64;
  const _Float16* Pb = P  + (size_t)b * HW * HW;
  const _Float16* Vb = Vh + (size_t)b * NC * HW;
  int  r  = lane & 15;
  bool lo = lane < 16;

  // Per-lane base pointers (k-invariant part hoisted out of the loop).
  const _Float16* prow0 = Pb + (size_t)(i0 +  0 + r) * HW + (lo ? 0 : 16);
  const _Float16* prow1 = Pb + (size_t)(i0 + 16 + r) * HW + (lo ? 0 : 16);
  const _Float16* vrow0 = Vb + (size_t)(c0 +  0 + r) * HW + (lo ? 0 : 16);
  const _Float16* vrow1 = Vb + (size_t)(c0 + 16 + r) * HW + (lo ? 0 : 16);
  const _Float16* vrow2 = Vb + (size_t)(c0 + 32 + r) * HW + (lo ? 0 : 16);
  const _Float16* vrow3 = Vb + (size_t)(c0 + 48 + r) * HW + (lo ? 0 : 16);

  v8f acc[2][4];
  for (int mt = 0; mt < 2; mt++) for (int nt = 0; nt < 4; nt++) { v8f z = {}; acc[mt][nt] = z; }

#pragma unroll 2
  for (int k0 = 0; k0 < HW; k0 += 32) {
    // A fragments (16x32 f16): lanes 0-15 row r hold K 0..15; lanes 16-31 hold K 16..31
    v16h a[2];
    {
      const _Float16* pr0 = prow0 + k0;
      const _Float16* pr1 = prow1 + k0;
      v8h a0l = *(const v8h*)pr0;
      v8h a0h = *(const v8h*)(pr0 + 8);
      v8h a1l = *(const v8h*)pr1;
      v8h a1h = *(const v8h*)(pr1 + 8);
      a[0] = __builtin_shufflevector(a0l, a0h, 0,1,2,3,4,5,6,7,8,9,10,11,12,13,14,15);
      a[1] = __builtin_shufflevector(a1l, a1h, 0,1,2,3,4,5,6,7,8,9,10,11,12,13,14,15);
    }
    // B fragments (32x16 f16): lane n<16 = column n, K 0..15; lane n>=16 = column n-16, K 16..31
    v16h bf[4];
    bf[0] = *(const v16h*)(vrow0 + k0);
    bf[1] = *(const v16h*)(vrow1 + k0);
    bf[2] = *(const v16h*)(vrow2 + k0);
    bf[3] = *(const v16h*)(vrow3 + k0);

    for (int mt = 0; mt < 2; mt++)
      for (int nt = 0; nt < 4; nt++)
        acc[mt][nt] = __builtin_amdgcn_wmma_f32_16x16x32_f16(
            false, a[mt], false, bf[nt], (short)0, acc[mt][nt], false, false);
  }

  // C/D layout: VGPR rr -> (M=rr, N=lane) for lanes 0-15, (M=rr+8, N=lane-16) for lanes 16-31
  float* ob = out + (size_t)b * HW * NC;
  int nb   = lane & 15;
  int mofs = lo ? 0 : 8;
  for (int mt = 0; mt < 2; mt++)
    for (int nt = 0; nt < 4; nt++)
      for (int rr = 0; rr < 8; rr++) {
        int i = i0 + mt*16 + rr + mofs;
        int c = c0 + nt*16 + nb;
        ob[(size_t)i * NC + c] = acc[mt][nt][rr];
      }
}

// ---------------- launch ----------------
extern "C" void kernel_launch(void* const* d_in, const int* in_sizes, int n_in,
                              void* d_out, int out_size, void* d_ws, size_t ws_size,
                              hipStream_t stream) {
  // f_tar (d_in[0]) is unused by the reference computation.
  const float* f_src = (const float*)d_in[1];
  const float* K1    = (const float*)d_in[2];
  const float* K2    = (const float*)d_in[3];
  const float* R     = (const float*)d_in[4];
  const float* t     = (const float*)d_in[5];

  // Workspace layout (~48.3 MB total):
  char* ws = (char*)d_ws;
  float4*   coefs  = (float4*)ws;                                   // 16*1024*16     =   262144 B
  float*    colsum = (float*)(ws + 262144);                         // 16*1024*4      =    65536 B
  _Float16* P      = (_Float16*)(ws + 262144 + 65536);              // 16*1024*1024*2 = 33554432 B
  _Float16* Vh     = (_Float16*)(ws + 262144 + 65536 + 33554432);   // 16*512*1024*2  = 16777216 B

  hipMemsetAsync(colsum, 0, NB * HW * sizeof(float), stream);
  k_setup     <<<NB, 64, 0, stream>>>(K1, K2, R, t, coefs);
  k_rowsoftmax<<<NB * HW, 256, 0, stream>>>(coefs, P);
  k_colsum    <<<NB * 4 * 8, 256, 0, stream>>>(P, colsum);
  k_buildV    <<<(NB * NC * HW / 4) / 256, 256, 0, stream>>>(f_src, colsum, Vh);
  k_gemm      <<<dim3(4, 8, NB), 256, 0, stream>>>(P, Vh, (float*)d_out);
}
